// MultiHeadAttention_6313601925858
// MI455X (gfx1250) — matrix operand order repaired and checked by reading, not compiled, not run on hardware
//
#include <hip/hip_runtime.h>
#include <hip/hip_bf16.h>

#define S_LEN 4096
#define DM    768
#define NH    12
#define DK    64

typedef __attribute__((ext_vector_type(16))) __bf16       v16bf;
typedef __attribute__((ext_vector_type(8)))  float        v8f;
typedef __attribute__((ext_vector_type(4)))  int          v4i;
typedef __attribute__((ext_vector_type(8)))  int          v8i;
typedef __attribute__((ext_vector_type(4)))  unsigned int v4u;
typedef __attribute__((ext_vector_type(2)))  unsigned int v2u;
typedef __attribute__((ext_vector_type(4)))  float        v4f;

union Frag16 { v4i i[2]; v16bf v; };

#if defined(__has_builtin)
#if __has_builtin(__builtin_amdgcn_tensor_load_to_lds)
#define HAVE_TDM 1
#endif
#endif
#ifndef HAVE_TDM
#define HAVE_TDM 0
#endif

__device__ __forceinline__ unsigned short f2bf(float f) {
  unsigned int u = __float_as_uint(f);
  u += 0x7FFFu + ((u >> 16) & 1u);                  // round-to-nearest-even
  return (unsigned short)(u >> 16);
}
__device__ __forceinline__ unsigned pack2(float a, float b) {
  return (unsigned)f2bf(a) | ((unsigned)f2bf(b) << 16);
}

// DPP16 lane permute (VALU, no LDS): CTRL must be an ICE -> template param.
template<int CTRL>
__device__ __forceinline__ float dpp_mov(float x) {
  return __int_as_float(__builtin_amdgcn_update_dpp(
      0, __float_as_int(x), CTRL, 0xF, 0xF, true));
}
// Butterfly reduction within each 16-lane half (C-fragment rows).
__device__ __forceinline__ float red_max16(float x) {
  x = fmaxf(x, dpp_mov<0x0B1>(x));   // xor1: quad_perm(1,0,3,2)
  x = fmaxf(x, dpp_mov<0x04E>(x));   // xor2: quad_perm(2,3,0,1)
  x = fmaxf(x, dpp_mov<0x141>(x));   // xor7: row_half_mirror
  x = fmaxf(x, dpp_mov<0x140>(x));   // xor15: row_mirror
  return x;
}
__device__ __forceinline__ float red_sum16(float x) {
  x += dpp_mov<0x0B1>(x);
  x += dpp_mov<0x04E>(x);
  x += dpp_mov<0x141>(x);
  x += dpp_mov<0x140>(x);
  return x;
}

#if HAVE_TDM
__device__ __forceinline__ unsigned lds_off_of(const void* p) {
  return (unsigned)(uintptr_t)p;                    // low 32 bits of generic = LDS offset
}
// TDM: 2-byte elements, tile tw x th, row stride in elems, optional LDS pad.
__device__ __forceinline__ void tdm_load_2d(const unsigned short* gsrc, unsigned ldsoff,
                                            unsigned tw, unsigned th, unsigned stride,
                                            unsigned pad_iv, unsigned pad_amt, unsigned pad_en)
{
  unsigned long long ga = (unsigned long long)(uintptr_t)gsrc;
  v4u g0;
  g0[0] = 1u;                                                   // count=1
  g0[1] = ldsoff;                                               // lds_addr
  g0[2] = (unsigned)ga;                                         // global_addr lo
  g0[3] = (unsigned)((ga >> 32) & 0x01FFFFFFull) | 0x80000000u; // addr[56:32] | type=2
  v8i g1;
  g1[0] = (int)((1u << 16) | (pad_en << 20) | (pad_iv << 22) | (pad_amt << 25));
  g1[1] = (int)((tw & 0xFFFFu) << 16);
  g1[2] = (int)(((tw >> 16) & 0xFFFFu) | ((th & 0xFFFFu) << 16));
  g1[3] = (int)(((th >> 16) & 0xFFFFu) | ((tw & 0xFFFFu) << 16));
  g1[4] = (int)(th & 0xFFFFu);
  g1[5] = (int)stride;
  g1[6] = 0;
  g1[7] = 0;
  v4i z = {0, 0, 0, 0};
#if defined(__clang_major__) && (__clang_major__ >= 23)
  v8i z8 = {0, 0, 0, 0, 0, 0, 0, 0};
  __builtin_amdgcn_tensor_load_to_lds(g0, g1, z, z, z8, 0);
#else
  __builtin_amdgcn_tensor_load_to_lds(g0, g1, z, z, 0);
#endif
}
#endif

// ---------------------------------------------------------------------------
__global__ __launch_bounds__(256)
void cvt_act_bf16(const float* __restrict__ in, unsigned short* __restrict__ out, int n4)
{
  int i = blockIdx.x * 256 + threadIdx.x;
  if (i < n4) {
    v4f d = *(const v4f*)(in + (size_t)i * 4);
    v2u p = { pack2(d.x, d.y), pack2(d.z, d.w) };
    *(v2u*)(out + (size_t)i * 4) = p;
  }
}

__global__ __launch_bounds__(256)
void cvt_wt_bf16(const float* __restrict__ W, unsigned short* __restrict__ Wt)
{
  __shared__ float tile[64][65];
  const int bx = blockIdx.x * 64;       // k base
  const int by = blockIdx.y * 64;       // n base
  const int c4 = (threadIdx.x & 15) * 4;
  const int r0 = threadIdx.x >> 4;
#pragma unroll
  for (int p = 0; p < 4; ++p) {
    int r = r0 + p * 16;
    v4f d = *(const v4f*)(W + (size_t)(bx + r) * DM + by + c4);
    tile[r][c4 + 0] = d.x; tile[r][c4 + 1] = d.y;
    tile[r][c4 + 2] = d.z; tile[r][c4 + 3] = d.w;
  }
  __syncthreads();
#pragma unroll
  for (int p = 0; p < 4; ++p) {
    int r = r0 + p * 16;
    v2u q = { pack2(tile[c4 + 0][r], tile[c4 + 1][r]),
              pack2(tile[c4 + 2][r], tile[c4 + 3][r]) };
    *(v2u*)(Wt + (size_t)(by + r) * DM + bx + c4) = q;
  }
}

// ---------------------------------------------------------------------------
__device__ __forceinline__ void mma_step(const unsigned short* __restrict__ As,
                                         const unsigned short* __restrict__ Bs,
                                         int wm, int wn, int lrow, int hiA, int kbB,
                                         v8f c[2][2])
{
  Frag16 a[2], b[2];
#pragma unroll
  for (int mi = 0; mi < 2; ++mi) {
    int m = wm + mi * 16 + lrow;
    a[mi].i[0] = *(const v4i*)&As[m * 40 + hiA];
    a[mi].i[1] = *(const v4i*)&As[m * 40 + 16 + hiA];
  }
#pragma unroll
  for (int ni = 0; ni < 2; ++ni) {
    int n = wn + ni * 16 + lrow;
    b[ni].i[0] = *(const v4i*)&Bs[n * 40 + (kbB >> 1)];
    b[ni].i[1] = *(const v4i*)&Bs[n * 40 + 16 + (kbB >> 1)];
  }
#pragma unroll
  for (int mi = 0; mi < 2; ++mi)
#pragma unroll
    for (int ni = 0; ni < 2; ++ni)
      c[mi][ni] = __builtin_amdgcn_wmma_f32_16x16x32_bf16(
          false, a[mi].v, false, b[ni].v, (short)0, c[mi][ni], false, false);
}

// ---------------------------------------------------------------------------
// GEMM: out[M,N] = Abf[M,768] @ Wt[N,768]^T + bias  (bf16 in, f32 accum)
// MODE 0: fp32 row-major. MODE 1: bf16 [H][S][DK] (*scl). MODE 2: bf16 [H][DK][S] (*scl).
// ---------------------------------------------------------------------------
template<int MODE>
__global__ __launch_bounds__(256)
void gemm_bf16wmma(const unsigned short* __restrict__ Abf,
                   const unsigned short* __restrict__ Wt,
                   const float* __restrict__ bias,
                   float* __restrict__ outF,
                   unsigned short* __restrict__ outBF,
                   float scl)
{
  __shared__ unsigned short As[2][128 * 40];
  __shared__ unsigned short Bs[2][64 * 40];

  const int tid = threadIdx.x, lane = tid & 31, wave = tid >> 5;
  const int wm = (wave & 3) * 32, wn = (wave >> 2) * 32;
  const int m0 = blockIdx.x * 128, n0 = blockIdx.y * 64;
  const int lrow = lane & 15;
  const int hiA = (lane >= 16) ? 8 : 0;
  const int kbB = (lane >= 16) ? 16 : 0;
  const int hiM = (lane >= 16) ? 8 : 0;
  constexpr int NT = DM / 32;

  v8f c[2][2] = {};

#if HAVE_TDM
  if (wave == 0) {   // prime buffer 0 (pad 4 dwords / 16 dwords -> LDS stride 40)
    tdm_load_2d(Abf + (size_t)m0 * DM, lds_off_of(&As[0][0]), 32, 128, DM, 3, 3, 1);
    tdm_load_2d(Wt  + (size_t)n0 * DM, lds_off_of(&Bs[0][0]), 32,  64, DM, 3, 3, 1);
  }
  for (int i = 0; i < NT; ++i) {
    const int buf = i & 1;
    if (wave == 0) __builtin_amdgcn_s_wait_tensorcnt(0);
    __syncthreads();
    if (wave == 0 && i + 1 < NT) {
      const int k0n = (i + 1) * 32;
      tdm_load_2d(Abf + (size_t)m0 * DM + k0n, lds_off_of(&As[buf ^ 1][0]), 32, 128, DM, 3, 3, 1);
      tdm_load_2d(Wt  + (size_t)n0 * DM + k0n, lds_off_of(&Bs[buf ^ 1][0]), 32,  64, DM, 3, 3, 1);
    }
    mma_step(&As[buf][0], &Bs[buf][0], wm, wn, lrow, hiA, kbB, c);
  }
#else
  for (int i = 0; i < NT; ++i) {
    const int k0 = i * 32;
    __syncthreads();
#pragma unroll
    for (int j = 0; j < 2; ++j) {
      int slot = tid + j * 256;
      int r = slot >> 2, seg = (slot & 3) << 3;
      *(v4i*)&As[0][r * 40 + seg] = *(const v4i*)(Abf + (size_t)(m0 + r) * DM + k0 + seg);
    }
    {
      int n = tid >> 2, seg = (tid & 3) << 3;
      *(v4i*)&Bs[0][n * 40 + seg] = *(const v4i*)(Wt + (size_t)(n0 + n) * DM + k0 + seg);
    }
    __syncthreads();
    mma_step(&As[0][0], &Bs[0][0], wm, wn, lrow, hiA, kbB, c);
  }
#endif

#pragma unroll
  for (int mi = 0; mi < 2; ++mi)
#pragma unroll
    for (int ni = 0; ni < 2; ++ni) {
      const int n = n0 + wn + ni * 16 + lrow;
      const float bn = bias[n];
#pragma unroll
      for (int r = 0; r < 8; ++r) {
        int m = m0 + wm + mi * 16 + r + hiM;
        float v = c[mi][ni][r] + bn;
        if (MODE == 0) {
          outF[(size_t)m * DM + n] = v;
        } else if (MODE == 1) {
          int h = n >> 6, d = n & 63;
          outBF[((size_t)h * S_LEN + m) * DK + d] = f2bf(v * scl);
        } else {   // MODE 2: transposed per-head [H][DK][S]
          int h = n >> 6, d = n & 63;
          outBF[((size_t)h * DK + d) * S_LEN + m] = f2bf(v * scl);
        }
      }
    }
}

// ---------------------------------------------------------------------------
// Causal flash attention. Q/K bf16 [H][S][64] (Q pre-scaled by 1/8),
// V bf16 [H][64][S] (pre-transposed). Out bf16 [S][768] head-concat.
// 4 waves x 16 queries; 32-key blocks; K+V via double-buffered TDM.
// ---------------------------------------------------------------------------
__global__ __launch_bounds__(128)
void flash_attn(const unsigned short* __restrict__ Q,
                const unsigned short* __restrict__ K,
                const unsigned short* __restrict__ V,
                unsigned short* __restrict__ Ob)
{
  __shared__ unsigned short Ks[2][32 * 64];    // [key][dk] = B of QK^T
  __shared__ unsigned short Vs[2][64 * 32];    // [dk][key] = B of P*V
  __shared__ unsigned short Ps[4][16 * 40];    // per-wave C->A layout scratch

  const int tid = threadIdx.x, lane = tid & 31, wave = tid >> 5;
  const int qb = blockIdx.x, h = blockIdx.y;

  const unsigned short* Qh = Q + (size_t)h * S_LEN * DK;
  const unsigned short* Kh = K + (size_t)h * S_LEN * DK;
  const unsigned short* Vh = V + (size_t)h * DK * S_LEN;   // rows = dk, stride S

  const int lrow = lane & 15;
  const int hiA = (lane >= 16) ? 8 : 0;
  const int kbB = (lane >= 16) ? 16 : 0;
  const int hiM = (lane >= 16) ? 8 : 0;
  const int qrow0 = qb * 64 + wave * 16;

  Frag16 qf[2];
  {
    int m = qrow0 + lrow;
#pragma unroll
    for (int j = 0; j < 2; ++j) {
      qf[j].i[0] = *(const v4i*)(Qh + (size_t)m * DK + j * 32 + hiA);
      qf[j].i[1] = *(const v4i*)(Qh + (size_t)m * DK + j * 32 + 16 + hiA);
    }
  }

  v8f o[4] = {};
  float mrow[8], lsum[8];
#pragma unroll
  for (int r = 0; r < 8; ++r) { mrow[r] = -1e30f; lsum[r] = 0.f; }

  const int nblk = 2 * qb + 2;

#if HAVE_TDM
  if (wave == 0) {   // prime buffer 0: K contiguous 4KB, V strided 32x64
    tdm_load_2d(Kh, lds_off_of(&Ks[0][0]), 2048, 1, 2048, 0, 0, 0);
    tdm_load_2d(Vh, lds_off_of(&Vs[0][0]), 32, 64, S_LEN, 0, 0, 0);
  }
#endif
  for (int kb = 0; kb < nblk; ++kb) {
    const int buf = kb & 1;
#if HAVE_TDM
    if (wave == 0) __builtin_amdgcn_s_wait_tensorcnt(0);
    __syncthreads();                       // publish tile kb; prior reads done
    if (wave == 0 && kb + 1 < nblk) {      // overlap DMA of tile kb+1
      tdm_load_2d(Kh + (size_t)(kb + 1) * 32 * DK, lds_off_of(&Ks[buf ^ 1][0]),
                  2048, 1, 2048, 0, 0, 0);
      tdm_load_2d(Vh + (size_t)(kb + 1) * 32, lds_off_of(&Vs[buf ^ 1][0]),
                  32, 64, S_LEN, 0, 0, 0);
    }
#else
    __syncthreads();
#pragma unroll
    for (int i = 0; i < 2; ++i) {
      int slot = tid + i * 128;
      *(v4i*)&Ks[buf][slot * 8] = *(const v4i*)(Kh + (size_t)kb * 32 * DK + slot * 8);
    }
#pragma unroll
    for (int i = 0; i < 2; ++i) {
      int slot = tid + i * 128;            // 256 slots: 64 rows x 4 segs
      int r = slot >> 2, seg = (slot & 3) << 3;
      *(v4i*)&Vs[buf][r * 32 + seg] =
          *(const v4i*)(Vh + (size_t)r * S_LEN + (size_t)kb * 32 + seg);
    }
    __syncthreads();
#endif

    // scores 16x32
    v8f s[2] = {};
#pragma unroll
    for (int t2 = 0; t2 < 2; ++t2) {
      int keyrow = t2 * 16 + lrow;
#pragma unroll
      for (int j = 0; j < 2; ++j) {
        Frag16 kf;
        kf.i[0] = *(const v4i*)&Ks[buf][keyrow * 64 + j * 32 + hiA];
        kf.i[1] = *(const v4i*)&Ks[buf][keyrow * 64 + j * 32 + 16 + hiA];
        s[t2] = __builtin_amdgcn_wmma_f32_16x16x32_bf16(
            false, qf[j].v, false, kf.v, (short)0, s[t2], false, false);
      }
    }

    // online softmax; reductions via DPP16 (pure VALU)
    const int k0g = kb * 32 + lrow;
#pragma unroll
    for (int r = 0; r < 8; ++r) {
      int mg = qrow0 + r + hiM;
      float s0 = (k0g      > mg) ? -1e9f : s[0][r];   // Q pre-scaled by 1/8
      float s1 = (k0g + 16 > mg) ? -1e9f : s[1][r];
      float rmax  = red_max16(fmaxf(s0, s1));
      float mnew  = fmaxf(mrow[r], rmax);
      float alpha = __expf(mrow[r] - mnew);
      mrow[r] = mnew;
      float p0 = __expf(s0 - mnew);
      float p1 = __expf(s1 - mnew);
      lsum[r] = lsum[r] * alpha + red_sum16(p0 + p1);
#pragma unroll
      for (int t4 = 0; t4 < 4; ++t4) o[t4][r] *= alpha;
      Ps[wave][(r + hiM) * 40 + lrow]      = f2bf(p0);
      Ps[wave][(r + hiM) * 40 + 16 + lrow] = f2bf(p1);
    }
    asm volatile("s_wait_dscnt 0" ::: "memory");

    Frag16 pf;
    pf.i[0] = *(const v4i*)&Ps[wave][lrow * 40 + hiA];
    pf.i[1] = *(const v4i*)&Ps[wave][lrow * 40 + 16 + hiA];
#pragma unroll
    for (int n4 = 0; n4 < 4; ++n4) {
      Frag16 vf;
      int dkr = n4 * 16 + lrow;
      vf.i[0] = *(const v4i*)&Vs[buf][dkr * 32 + hiA];
      vf.i[1] = *(const v4i*)&Vs[buf][dkr * 32 + 16 + hiA];
      o[n4] = __builtin_amdgcn_wmma_f32_16x16x32_bf16(
          false, pf.v, false, vf.v, (short)0, o[n4], false, false);
    }
  }

  float rl[8];
#pragma unroll
  for (int r = 0; r < 8; ++r) rl[r] = 1.0f / lsum[r];
#pragma unroll
  for (int n4 = 0; n4 < 4; ++n4)
#pragma unroll
    for (int r = 0; r < 8; ++r) {
      int m = qrow0 + r + hiM;
      int d = n4 * 16 + lrow;
      Ob[(size_t)m * DM + h * DK + d] = f2bf(o[n4][r] * rl[r]);
    }
}

// ---------------------------------------------------------------------------
extern "C" void kernel_launch(void* const* d_in, const int* in_sizes, int n_in,
                              void* d_out, int out_size, void* d_ws, size_t ws_size,
                              hipStream_t stream)
{
  (void)in_sizes; (void)n_in; (void)out_size; (void)ws_size;
  const float* q  = (const float*)d_in[0];
  const float* k  = (const float*)d_in[1];
  const float* v  = (const float*)d_in[2];
  // d_in[3] = causal mask: handled analytically
  const float* Wq = (const float*)d_in[4];
  const float* bq = (const float*)d_in[5];
  const float* Wk = (const float*)d_in[6];
  const float* bk = (const float*)d_in[7];
  const float* Wv = (const float*)d_in[8];
  const float* bv = (const float*)d_in[9];
  const float* Wo = (const float*)d_in[10];
  const float* bo = (const float*)d_in[11];
  float* out = (float*)d_out;

  const size_t SD = (size_t)S_LEN * DM;
  const size_t WW = (size_t)DM * DM;
  unsigned short* xq  = (unsigned short*)d_ws;
  unsigned short* xk  = xq  + SD;
  unsigned short* xv  = xk  + SD;
  unsigned short* WqT = xv  + SD;
  unsigned short* WkT = WqT + WW;
  unsigned short* WvT = WkT + WW;
  unsigned short* WoT = WvT + WW;
  unsigned short* qbf = WoT + WW;           // [H][S][64], pre-scaled 1/8
  unsigned short* kbf = qbf + SD;           // [H][S][64]
  unsigned short* vbf = kbf + SD;           // [H][64][S] (transposed)
  unsigned short* attnbf = xq;              // xq dead after Q projection

  const int n4 = (int)(SD / 4);
  cvt_act_bf16<<<(n4 + 255) / 256, 256, 0, stream>>>(q, xq, n4);
  cvt_act_bf16<<<(n4 + 255) / 256, 256, 0, stream>>>(k, xk, n4);
  cvt_act_bf16<<<(n4 + 255) / 256, 256, 0, stream>>>(v, xv, n4);
  dim3 gw(DM / 64, DM / 64);
  cvt_wt_bf16<<<gw, 256, 0, stream>>>(Wq, WqT);
  cvt_wt_bf16<<<gw, 256, 0, stream>>>(Wk, WkT);
  cvt_wt_bf16<<<gw, 256, 0, stream>>>(Wv, WvT);
  cvt_wt_bf16<<<gw, 256, 0, stream>>>(Wo, WoT);

  dim3 gg(S_LEN / 128, DM / 64);
  gemm_bf16wmma<1><<<gg, 256, 0, stream>>>(xq, WqT, bq, nullptr, qbf, 0.125f);
  gemm_bf16wmma<1><<<gg, 256, 0, stream>>>(xk, WkT, bk, nullptr, kbf, 1.0f);
  gemm_bf16wmma<2><<<gg, 256, 0, stream>>>(xv, WvT, bv, nullptr, vbf, 1.0f);
  flash_attn<<<dim3(S_LEN / 64, NH), 128, 0, stream>>>(qbf, kbf, vbf, attnbf);
  gemm_bf16wmma<0><<<gg, 256, 0, stream>>>(attnbf, WoT, bo, out, nullptr, 1.0f);
}
